// PointBertEncoder_69544110457439
// MI455X (gfx1250) — matrix operand order repaired and checked by reading, not compiled, not run on hardware
//
#include <hip/hip_runtime.h>
#include <cstdint>
#include <cstddef>

// ---------------------------------------------------------------------------
// PointBERT encoder for MI455X (gfx1250, wave32, WMMA).
// Input flattening assumption (jax pytree: dict keys sorted at every level):
//  0:blocks.b1  1:blocks.b2  2:blocks.bo  3:blocks.bqkv  4:ln1_b 5:ln1_g
//  6:ln2_b 7:ln2_g 8:blocks.w1 9:blocks.w2 10:blocks.wo 11:blocks.wqkv
//  12:cls 13:norm_b 14:norm_g
//  15..32: pn.{b1,b2,b3,be1,be2,be3,g1,g2,g3,m1,m2,m3,v1,v2,v3,w1,w2,w3}
//  33:pos 34:points
// ---------------------------------------------------------------------------

#define B_   32
#define N_   8192
#define D_   384
#define L_   12
#define H_   6
#define G_   512
#define K_   32
#define MLP_ 1536
#define SP_  544              // 513 padded to multiple of 32
#define DR_  (B_ * SP_)       // 17408 activation rows
#define QKVW_ 1152
#define EPS_ 1e-5f

typedef __attribute__((ext_vector_type(16))) _Float16 v16h;
typedef __attribute__((ext_vector_type(8)))  float    v8f;

union Frag {
    uint4     u[2];
    _Float16  h[16];
    v16h      v;
};

__device__ __forceinline__ v8f wmma16(const Frag& a, const Frag& b, v8f c) {
    // D = A(16x32 f16) * B(32x16 f16) + C(16x16 f32)
    return __builtin_amdgcn_wmma_f32_16x16x32_f16(
        /*neg_a=*/false, a.v, /*neg_b=*/false, b.v,
        /*c_mod=*/(short)0, c, /*reuse_a=*/false, /*reuse_b=*/false);
}

__device__ __forceinline__ float gelu_exact(float x) {
    return 0.5f * x * (1.0f + erff(x * 0.70710678118654752f));
}

// --------------------------- f32 -> f16 convert ----------------------------
__global__ void f2h_kernel(const float* __restrict__ in, _Float16* __restrict__ out, size_t n) {
    size_t i = (size_t)blockIdx.x * blockDim.x + threadIdx.x;
    size_t stride = (size_t)gridDim.x * blockDim.x;
    for (; i < n; i += stride) out[i] = (_Float16)in[i];
}

// ------------------------------- FPS ---------------------------------------
// One block (256 threads) per batch; 512 sequential argmax iterations.
__global__ void fps_kernel(const float* __restrict__ points, float* __restrict__ centers) {
    __shared__ float sv[256];
    __shared__ int   si[256];
    __shared__ int   sfar;
    const int b = blockIdx.x;
    const int t = threadIdx.x;
    const float* P = points + (size_t)b * N_ * 3;

    float dist[N_ / 256];
#pragma unroll
    for (int j = 0; j < N_ / 256; ++j) dist[j] = 1e10f;
    if (t == 0) sfar = 0;
    __syncthreads();

    for (int n = 0; n < G_; ++n) {
        const int far = sfar;
        const float cx = P[far * 3 + 0], cy = P[far * 3 + 1], cz = P[far * 3 + 2];
        if (t == 0) {
            centers[((size_t)b * G_ + n) * 3 + 0] = cx;
            centers[((size_t)b * G_ + n) * 3 + 1] = cy;
            centers[((size_t)b * G_ + n) * 3 + 2] = cz;
        }
        float bv = -1.0f; int bi = 0;
#pragma unroll
        for (int j = 0; j < N_ / 256; ++j) {
            const int p = t + j * 256;
            const float dx = P[p * 3 + 0] - cx, dy = P[p * 3 + 1] - cy, dz = P[p * 3 + 2] - cz;
            const float d = dx * dx + dy * dy + dz * dz;
            dist[j] = fminf(dist[j], d);
            if (dist[j] > bv || (dist[j] == bv && p < bi)) { bv = dist[j]; bi = p; }
        }
        sv[t] = bv; si[t] = bi;
        __syncthreads();
        for (int off = 128; off > 0; off >>= 1) {
            if (t < off) {
                if (sv[t + off] > sv[t] || (sv[t + off] == sv[t] && si[t + off] < si[t])) {
                    sv[t] = sv[t + off]; si[t] = si[t + off];
                }
            }
            __syncthreads();
        }
        if (t == 0) sfar = si[0];
        __syncthreads();
    }
}

// ------------------------------- kNN ---------------------------------------
// One wave (32 threads) per (b,g) center; top-32 by repeated wave-min.
__global__ void knn_kernel(const float* __restrict__ points,
                           const float* __restrict__ centers,
                           int* __restrict__ knn) {
    __shared__ float sd[N_];
    const int bg = blockIdx.x;
    const int b  = bg >> 9;
    const int lane = threadIdx.x;
    const float cx = centers[bg * 3 + 0], cy = centers[bg * 3 + 1], cz = centers[bg * 3 + 2];
    const float* P = points + (size_t)b * N_ * 3;

    for (int p = lane; p < N_; p += 32) {
        const float dx = P[p * 3 + 0] - cx, dy = P[p * 3 + 1] - cy, dz = P[p * 3 + 2] - cz;
        sd[p] = dx * dx + dy * dy + dz * dz;
    }
    for (int r = 0; r < K_; ++r) {
        float bv = 1e30f; int bi = 0x7fffffff;
        for (int j = 0; j < N_ / 32; ++j) {
            const int p = lane + j * 32;
            const float d = sd[p];
            if (d < bv || (d == bv && p < bi)) { bv = d; bi = p; }
        }
#pragma unroll
        for (int m = 1; m < 32; m <<= 1) {
            const float ov = __shfl_xor(bv, m);
            const int   oi = __shfl_xor(bi, m);
            if (ov < bv || (ov == bv && oi < bi)) { bv = ov; bi = oi; }
        }
        if ((bi & 31) == lane) sd[bi] = 1e30f;   // owner lane invalidates
        if (lane == 0) knn[(size_t)bg * K_ + r] = bi;
    }
}

// ---------------------------- PointNet MLP ---------------------------------
// One block (256 threads) per (b,g). 3->64->128->384 BN(eval)+ReLU, max over K.
__global__ void pointnet_kernel(const float* __restrict__ points,
                                const float* __restrict__ centers,
                                const int* __restrict__ knn,
                                const float* __restrict__ w1, const float* __restrict__ b1,
                                const float* __restrict__ g1, const float* __restrict__ be1,
                                const float* __restrict__ m1, const float* __restrict__ v1,
                                const float* __restrict__ w2, const float* __restrict__ b2,
                                const float* __restrict__ g2, const float* __restrict__ be2,
                                const float* __restrict__ m2, const float* __restrict__ v2,
                                const float* __restrict__ w3, const float* __restrict__ b3,
                                const float* __restrict__ g3, const float* __restrict__ be3,
                                const float* __restrict__ m3, const float* __restrict__ v3,
                                float* __restrict__ X) {
    __shared__ float rp[K_ * 3];
    __shared__ float h1s[K_ * 64];
    __shared__ float h2s[K_ * 128];
    const int bg = blockIdx.x;
    const int b  = bg >> 9;
    const int gi = bg & 511;
    const int t  = threadIdx.x;

    if (t < K_) {
        const int pi = knn[(size_t)bg * K_ + t];
        const float* p = points + ((size_t)b * N_ + pi) * 3;
        rp[t * 3 + 0] = p[0] - centers[bg * 3 + 0];
        rp[t * 3 + 1] = p[1] - centers[bg * 3 + 1];
        rp[t * 3 + 2] = p[2] - centers[bg * 3 + 2];
    }
    __syncthreads();
    {
        const int k = t >> 3, o0 = (t & 7) * 8;
        const float px = rp[k * 3 + 0], py = rp[k * 3 + 1], pz = rp[k * 3 + 2];
        for (int o = o0; o < o0 + 8; ++o) {
            float s = w1[o * 3 + 0] * px + w1[o * 3 + 1] * py + w1[o * 3 + 2] * pz + b1[o];
            s = (s - m1[o]) * rsqrtf(v1[o] + EPS_) * g1[o] + be1[o];
            h1s[k * 64 + o] = fmaxf(s, 0.0f);
        }
    }
    __syncthreads();
    {
        const int k = t >> 3, o0 = (t & 7) * 16;
        for (int o = o0; o < o0 + 16; ++o) {
            float s = b2[o];
            const float* w = w2 + (size_t)o * 64;
            for (int j = 0; j < 64; ++j) s += h1s[k * 64 + j] * w[j];
            s = (s - m2[o]) * rsqrtf(v2[o] + EPS_) * g2[o] + be2[o];
            h2s[k * 128 + o] = fmaxf(s, 0.0f);
        }
    }
    __syncthreads();
    for (int c = t; c < D_; c += 256) {
        const float sc  = rsqrtf(v3[c] + EPS_) * g3[c];
        const float sh  = be3[c] - m3[c] * sc;
        const float* w  = w3 + (size_t)c * 128;
        float mx = -1e30f;
        for (int k = 0; k < K_; ++k) {
            float s = b3[c];
            for (int j = 0; j < 128; ++j) s += h2s[k * 128 + j] * w[j];
            mx = fmaxf(mx, s * sc + sh);
        }
        X[((size_t)b * SP_ + 1 + gi) * D_ + c] = mx;
    }
}

// ---------------------- assemble X: cls + pos, zero pad ---------------------
__global__ void assemble_kernel(const float* __restrict__ cls, const float* __restrict__ pos,
                                float* __restrict__ X) {
    const size_t i = (size_t)blockIdx.x * blockDim.x + threadIdx.x;
    if (i >= (size_t)DR_ * D_) return;
    const int d = (int)(i % D_);
    const size_t row = i / D_;
    const int s = (int)(row % SP_);
    float v;
    if (s == 0)        v = cls[d] + pos[d];
    else if (s < 513)  v = X[i] + pos[(size_t)s * D_ + d];
    else               v = 0.0f;
    X[i] = v;
}

// --------------------------- LayerNorm -> f16 -------------------------------
__global__ void ln_f16_kernel(const float* __restrict__ X, const float* __restrict__ g,
                              const float* __restrict__ bb, _Float16* __restrict__ out, int rows) {
    const int lane = threadIdx.x & 31;
    const int row  = (int)((blockIdx.x * blockDim.x + threadIdx.x) >> 5);
    if (row >= rows) return;
    const float* x = X + (size_t)row * D_;
    float s = 0.f, s2 = 0.f;
#pragma unroll
    for (int t = 0; t < D_ / 32; ++t) { const float v = x[lane + 32 * t]; s += v; s2 += v * v; }
#pragma unroll
    for (int m = 1; m < 32; m <<= 1) { s += __shfl_xor(s, m); s2 += __shfl_xor(s2, m); }
    const float mu = s * (1.0f / D_);
    const float var = s2 * (1.0f / D_) - mu * mu;
    const float rs = rsqrtf(var + EPS_);
#pragma unroll
    for (int t = 0; t < D_ / 32; ++t) {
        const int d = lane + 32 * t;
        out[(size_t)row * D_ + d] = (_Float16)((x[d] - mu) * rs * g[d] + bb[d]);
    }
}

// ----------------------- final LayerNorm -> f32 out -------------------------
__global__ void ln_final_kernel(const float* __restrict__ X, const float* __restrict__ g,
                                const float* __restrict__ bb, float* __restrict__ out) {
    const int lane = threadIdx.x & 31;
    const int row  = (int)((blockIdx.x * blockDim.x + threadIdx.x) >> 5);
    if (row >= B_ * 513) return;
    const int b = row / 513, s = row % 513;
    const float* x = X + ((size_t)b * SP_ + s) * D_;
    float su = 0.f, s2 = 0.f;
#pragma unroll
    for (int t = 0; t < D_ / 32; ++t) { const float v = x[lane + 32 * t]; su += v; s2 += v * v; }
#pragma unroll
    for (int m = 1; m < 32; m <<= 1) { su += __shfl_xor(su, m); s2 += __shfl_xor(s2, m); }
    const float mu = su * (1.0f / D_);
    const float var = s2 * (1.0f / D_) - mu * mu;
    const float rs = rsqrtf(var + EPS_);
#pragma unroll
    for (int t = 0; t < D_ / 32; ++t) {
        const int d = lane + 32 * t;
        out[(size_t)row * D_ + d] = (x[d] - mu) * rs * g[d] + bb[d];
    }
}

// --------------------- generic WMMA GEMM: C = A * W^T + bias ----------------
// A: (M,K) f16 row-major; W: (N,K) f16 row-major.
// One wave computes a 16x64 output tile (A fragment reused across 4 B frags):
// 4 WMMA per K-step for 10 b128 loads -> 1.6x the matrix-issue density of a
// 16x16 tiling, and 4x fewer A refetches. N must be a multiple of 64.
// MODE 0: store f32   MODE 1: f32 += (residual)   MODE 2: store f16
// ACT  0: none        ACT  1: exact GELU
template <int MODE, int ACT>
__global__ void gemm_f16(const _Float16* __restrict__ A, const _Float16* __restrict__ W,
                         const float* __restrict__ bias, float* __restrict__ Cf,
                         _Float16* __restrict__ Ch, int M, int N, int K) {
    const int lane = threadIdx.x & 31;
    const int wid  = (int)((blockIdx.x * blockDim.x + threadIdx.x) >> 5);
    const int tn = N >> 6;                     // 64-wide N tiles
    const int tiles = (M >> 4) * tn;
    if (wid >= tiles) return;
    const int tm = wid / tn, tj = wid % tn;
    const int g = lane >> 4, ln = lane & 15;
    const _Float16* arow = A + (size_t)(tm * 16 + ln) * K;
    const _Float16* wrow0 = W + (size_t)(tj * 64 + ln) * K;          // cols +0..15
    const _Float16* wrow1 = wrow0 + (size_t)16 * K;                  // cols +16..31
    const _Float16* wrow2 = wrow0 + (size_t)32 * K;                  // cols +32..47
    const _Float16* wrow3 = wrow0 + (size_t)48 * K;                  // cols +48..63
    v8f acc[4];
#pragma unroll
    for (int c = 0; c < 4; ++c) acc[c] = (v8f){0.f, 0.f, 0.f, 0.f, 0.f, 0.f, 0.f, 0.f};

    for (int kk = 0; kk < K; kk += 32) {
        Frag fa, fb0, fb1, fb2, fb3;
        // A fragment: lane=row, halves j -> k = kk + 16g + j (contiguous 16)
        fa.u[0] = *(const uint4*)(arow + kk + 16 * g);
        fa.u[1] = *(const uint4*)(arow + kk + 16 * g + 8);
        // B fragments: lane=col, halves j -> k = kk + 16g + j (contiguous 16)
        fb0.u[0] = *(const uint4*)(wrow0 + kk + 16 * g);
        fb0.u[1] = *(const uint4*)(wrow0 + kk + 16 * g + 8);
        fb1.u[0] = *(const uint4*)(wrow1 + kk + 16 * g);
        fb1.u[1] = *(const uint4*)(wrow1 + kk + 16 * g + 8);
        fb2.u[0] = *(const uint4*)(wrow2 + kk + 16 * g);
        fb2.u[1] = *(const uint4*)(wrow2 + kk + 16 * g + 8);
        fb3.u[0] = *(const uint4*)(wrow3 + kk + 16 * g);
        fb3.u[1] = *(const uint4*)(wrow3 + kk + 16 * g + 8);
        if (kk + 64 <= K) {                    // prefetch next K slab of W
            __builtin_prefetch(wrow0 + kk + 32, 0, 1);
            __builtin_prefetch(wrow2 + kk + 32, 0, 1);
        }
        acc[0] = wmma16(fa, fb0, acc[0]);
        acc[1] = wmma16(fa, fb1, acc[1]);
        acc[2] = wmma16(fa, fb2, acc[2]);
        acc[3] = wmma16(fa, fb3, acc[3]);
    }
#pragma unroll
    for (int c = 0; c < 4; ++c) {
        const int col = tj * 64 + c * 16 + ln;
        const float bv = bias ? bias[col] : 0.0f;
#pragma unroll
        for (int r = 0; r < 8; ++r) {
            const int row = tm * 16 + r + 8 * g;
            float v = acc[c][r] + bv;
            if (ACT == 1) v = gelu_exact(v);
            const size_t idx = (size_t)row * N + col;
            if (MODE == 0)      Cf[idx] = v;
            else if (MODE == 1) Cf[idx] += v;
            else                Ch[idx] = (_Float16)v;
        }
    }
}

// ------------------------ flash attention (WMMA) ----------------------------
// qkv: (DR_, 1152) f16, [q|k|v] each H*64 wide. One wave per (b,h, 16-row q tile).
__global__ void attn_kernel(const _Float16* __restrict__ qkv, float* __restrict__ Y) {
    __shared__ __align__(16) _Float16 pst[8][16 * 32];   // per-wave P staging tile
    const int wib  = threadIdx.x >> 5;
    const int lane = threadIdx.x & 31;
    const int wid  = (int)(blockIdx.x * (blockDim.x >> 5) + wib);
    const int QT = SP_ / 16;                              // 34
    if (wid >= B_ * H_ * QT) return;
    const int qt = wid % QT;
    const int h  = (wid / QT) % H_;
    const int b  = wid / (QT * H_);
    const int g  = lane >> 4, ln = lane & 15;
    const _Float16* base = qkv + (size_t)b * SP_ * QKVW_;

    // Q A-fragments for k-steps 0 and 32 (HD = 64)
    const _Float16* qrow = base + (size_t)(qt * 16 + ln) * QKVW_ + h * 64;
    Frag aq0, aq1;
    aq0.u[0] = *(const uint4*)(qrow + 0 + 16 * g);
    aq0.u[1] = *(const uint4*)(qrow + 8 + 16 * g);
    aq1.u[0] = *(const uint4*)(qrow + 32 + 16 * g);
    aq1.u[1] = *(const uint4*)(qrow + 40 + 16 * g);

    float mi[8], li[8];
    v8f O0 = {0,0,0,0,0,0,0,0}, O1 = O0, O2 = O0, O3 = O0;
#pragma unroll
    for (int r = 0; r < 8; ++r) { mi[r] = -1e30f; li[r] = 0.0f; }

    for (int c = 0; c < SP_ / 32; ++c) {                  // 17 kv chunks of 32
        v8f sc[2];
#pragma unroll
        for (int t = 0; t < 2; ++t) {
            const int kv0 = c * 32 + t * 16;
            const _Float16* krow = base + (size_t)(kv0 + ln) * QKVW_ + D_ + h * 64;
            Frag bk0, bk1;
            bk0.u[0] = *(const uint4*)(krow + 0 + 16 * g);
            bk0.u[1] = *(const uint4*)(krow + 8 + 16 * g);
            bk1.u[0] = *(const uint4*)(krow + 32 + 16 * g);
            bk1.u[1] = *(const uint4*)(krow + 40 + 16 * g);
            v8f s = {0,0,0,0,0,0,0,0};
            s = wmma16(aq0, bk0, s);
            s = wmma16(aq1, bk1, s);
            const bool oob = (kv0 + ln) >= 513;           // mask padded kv columns
#pragma unroll
            for (int r = 0; r < 8; ++r) sc[t][r] = oob ? -1e30f : s[r] * 0.125f;
        }
        // online softmax row statistics (row spread over a 16-lane half-group)
        float p0[8], p1[8];
#pragma unroll
        for (int r = 0; r < 8; ++r) {
            float mx = fmaxf(sc[0][r], sc[1][r]);
#pragma unroll
            for (int m = 1; m < 16; m <<= 1) mx = fmaxf(mx, __shfl_xor(mx, m));
            const float nm = fmaxf(mi[r], mx);
            const float al = __expf(mi[r] - nm);
            const float e0 = __expf(sc[0][r] - nm);
            const float e1 = __expf(sc[1][r] - nm);
            float rs = e0 + e1;
#pragma unroll
            for (int m = 1; m < 16; m <<= 1) rs += __shfl_xor(rs, m);
            li[r] = li[r] * al + rs;
            mi[r] = nm;
            O0[r] *= al; O1[r] *= al; O2[r] *= al; O3[r] *= al;
            p0[r] = e0; p1[r] = e1;
        }
        // transpose P (C-layout, lane=col) -> A-layout (lane=row) via LDS
#pragma unroll
        for (int r = 0; r < 8; ++r) {
            const int row = r + 8 * g;
            pst[wib][row * 32 + ln]      = (_Float16)p0[r];
            pst[wib][row * 32 + 16 + ln] = (_Float16)p1[r];
        }
        Frag pa;
        const _Float16* pr = &pst[wib][ln * 32 + 16 * g];
        pa.u[0] = *(const uint4*)(pr);
        pa.u[1] = *(const uint4*)(pr + 8);
        // O += P(16x32) * V(32x64)
        const int cb = c * 32;
#pragma unroll
        for (int dt = 0; dt < 4; ++dt) {
            Frag bv;
#pragma unroll
            for (int j = 0; j < 16; ++j)
                bv.h[j] = base[(size_t)(cb + 16 * g + j) * QKVW_ + 2 * D_ + h * 64 + dt * 16 + ln];
            v8f* O = (dt == 0) ? &O0 : (dt == 1) ? &O1 : (dt == 2) ? &O2 : &O3;
            *O = wmma16(pa, bv, *O);
        }
    }
    // epilogue: normalize and scatter to Y (DR_, 384)
#pragma unroll
    for (int dt = 0; dt < 4; ++dt) {
        const v8f& O = (dt == 0) ? O0 : (dt == 1) ? O1 : (dt == 2) ? O2 : O3;
        const int col = h * 64 + dt * 16 + ln;
#pragma unroll
        for (int r = 0; r < 8; ++r) {
            const int row = qt * 16 + r + 8 * g;
            Y[(size_t)(b * SP_ + row) * D_ + col] = O[r] / li[r];
        }
    }
}

// ---------------------------------------------------------------------------
extern "C" void kernel_launch(void* const* d_in, const int* in_sizes, int n_in,
                              void* d_out, int out_size, void* d_ws, size_t ws_size,
                              hipStream_t stream) {
    (void)in_sizes; (void)n_in; (void)out_size; (void)ws_size;

    const float* blk_b1   = (const float*)d_in[0];
    const float* blk_b2   = (const float*)d_in[1];
    const float* blk_bo   = (const float*)d_in[2];
    const float* blk_bqkv = (const float*)d_in[3];
    const float* ln1_b    = (const float*)d_in[4];
    const float* ln1_g    = (const float*)d_in[5];
    const float* ln2_b    = (const float*)d_in[6];
    const float* ln2_g    = (const float*)d_in[7];
    const float* blk_w1   = (const float*)d_in[8];
    const float* blk_w2   = (const float*)d_in[9];
    const float* blk_wo   = (const float*)d_in[10];
    const float* blk_wqkv = (const float*)d_in[11];
    const float* cls      = (const float*)d_in[12];
    const float* norm_b   = (const float*)d_in[13];
    const float* norm_g   = (const float*)d_in[14];
    const float* pn_b1  = (const float*)d_in[15];
    const float* pn_b2  = (const float*)d_in[16];
    const float* pn_b3  = (const float*)d_in[17];
    const float* pn_be1 = (const float*)d_in[18];
    const float* pn_be2 = (const float*)d_in[19];
    const float* pn_be3 = (const float*)d_in[20];
    const float* pn_g1  = (const float*)d_in[21];
    const float* pn_g2  = (const float*)d_in[22];
    const float* pn_g3  = (const float*)d_in[23];
    const float* pn_m1  = (const float*)d_in[24];
    const float* pn_m2  = (const float*)d_in[25];
    const float* pn_m3  = (const float*)d_in[26];
    const float* pn_v1  = (const float*)d_in[27];
    const float* pn_v2  = (const float*)d_in[28];
    const float* pn_v3  = (const float*)d_in[29];
    const float* pn_w1  = (const float*)d_in[30];
    const float* pn_w2  = (const float*)d_in[31];
    const float* pn_w3  = (const float*)d_in[32];
    const float* pos    = (const float*)d_in[33];
    const float* points = (const float*)d_in[34];

    // -------- workspace carve (≈206 MB) --------
    size_t off = 0;
    char* wsb = (char*)d_ws;
    auto carve = [&](size_t bytes) -> char* {
        char* p = wsb + off;
        off += (bytes + 255) & ~(size_t)255;
        return p;
    };
    float*    centers = (float*)carve((size_t)B_ * G_ * 3 * 4);
    int*      knn     = (int*)  carve((size_t)B_ * G_ * K_ * 4);
    float*    X       = (float*)carve((size_t)DR_ * D_ * 4);
    float*    Yb      = (float*)carve((size_t)DR_ * D_ * 4);
    _Float16* Xh      = (_Float16*)carve((size_t)DR_ * D_ * 2);
    _Float16* QKVh    = (_Float16*)carve((size_t)DR_ * QKVW_ * 2);
    _Float16* Hh      = (_Float16*)carve((size_t)DR_ * MLP_ * 2);
    _Float16* Wh      = (_Float16*)carve((size_t)21233664 * 2);

    const size_t nWqkv = (size_t)L_ * 3 * D_ * D_;   // 5,308,416
    const size_t nWo   = (size_t)L_ * D_ * D_;       // 1,769,472
    const size_t nW1   = (size_t)L_ * MLP_ * D_;     // 7,077,888
    const size_t nW2   = (size_t)L_ * D_ * MLP_;     // 7,077,888
    _Float16* Wqkv_h = Wh;
    _Float16* Wo_h   = Wh + nWqkv;
    _Float16* W1_h   = Wo_h + nWo;
    _Float16* W2_h   = W1_h + nW1;

    auto cgrid = [](size_t n) { return (unsigned)((n + 255) / 256); };

    // weights -> f16 (once per call; deterministic)
    f2h_kernel<<<cgrid(nWqkv), 256, 0, stream>>>(blk_wqkv, Wqkv_h, nWqkv);
    f2h_kernel<<<cgrid(nWo),   256, 0, stream>>>(blk_wo,   Wo_h,   nWo);
    f2h_kernel<<<cgrid(nW1),   256, 0, stream>>>(blk_w1,   W1_h,   nW1);
    f2h_kernel<<<cgrid(nW2),   256, 0, stream>>>(blk_w2,   W2_h,   nW2);

    // geometry pipeline
    fps_kernel<<<B_, 256, 0, stream>>>(points, centers);
    knn_kernel<<<B_ * G_, 32, 0, stream>>>(points, centers, knn);
    pointnet_kernel<<<B_ * G_, 256, 0, stream>>>(points, centers, knn,
        pn_w1, pn_b1, pn_g1, pn_be1, pn_m1, pn_v1,
        pn_w2, pn_b2, pn_g2, pn_be2, pn_m2, pn_v2,
        pn_w3, pn_b3, pn_g3, pn_be3, pn_m3, pn_v3, X);
    assemble_kernel<<<cgrid((size_t)DR_ * D_), 256, 0, stream>>>(cls, pos, X);

    auto gemm_grid = [](int M, int N) { return (unsigned)(((size_t)(M / 16) * (N / 64) + 7) / 8); };
    const unsigned lnGrid   = DR_ / 8;                    // 8 rows per 256-thread block
    const unsigned attnGrid = (B_ * H_ * (SP_ / 16)) / 8; // 816

    for (int l = 0; l < L_; ++l) {
        // LN1 -> f16
        ln_f16_kernel<<<lnGrid, 256, 0, stream>>>(X, ln1_g + (size_t)l * D_, ln1_b + (size_t)l * D_, Xh, DR_);
        // QKV = Xh * Wqkv^T + bqkv   (f16 out)
        gemm_f16<2, 0><<<gemm_grid(DR_, 3 * D_), 256, 0, stream>>>(
            Xh, Wqkv_h + (size_t)l * 3 * D_ * D_, blk_bqkv + (size_t)l * 3 * D_,
            nullptr, QKVh, DR_, 3 * D_, D_);
        // attention -> Yb (f32)
        attn_kernel<<<attnGrid, 256, 0, stream>>>(QKVh, Yb);
        // Yb -> f16
        f2h_kernel<<<cgrid((size_t)DR_ * D_), 256, 0, stream>>>(Yb, Xh, (size_t)DR_ * D_);
        // X += Yh * Wo^T + bo
        gemm_f16<1, 0><<<gemm_grid(DR_, D_), 256, 0, stream>>>(
            Xh, Wo_h + (size_t)l * D_ * D_, blk_bo + (size_t)l * D_,
            X, nullptr, DR_, D_, D_);
        // LN2 -> f16
        ln_f16_kernel<<<lnGrid, 256, 0, stream>>>(X, ln2_g + (size_t)l * D_, ln2_b + (size_t)l * D_, Xh, DR_);
        // H = gelu(Xh * W1^T + b1)  (f16 out)
        gemm_f16<2, 1><<<gemm_grid(DR_, MLP_), 256, 0, stream>>>(
            Xh, W1_h + (size_t)l * MLP_ * D_, blk_b1 + (size_t)l * MLP_,
            nullptr, Hh, DR_, MLP_, D_);
        // X += H * W2^T + b2
        gemm_f16<1, 0><<<gemm_grid(DR_, D_), 256, 0, stream>>>(
            Hh, W2_h + (size_t)l * D_ * MLP_, blk_b2 + (size_t)l * D_,
            X, nullptr, DR_, D_, MLP_);
    }

    // final LN -> d_out (B,513,D) f32
    ln_final_kernel<<<(B_ * 513) / 8, 256, 0, stream>>>(X, norm_g, norm_b, (float*)d_out);
}
